// TransformerBlock_14671608283833
// MI455X (gfx1250) — matrix operand order repaired and checked by reading, not compiled, not run on hardware
//
#include <hip/hip_runtime.h>
#include <math.h>

typedef _Float16 v16h __attribute__((ext_vector_type(16)));
typedef _Float16 v8h  __attribute__((ext_vector_type(8)));
typedef float    v8f  __attribute__((ext_vector_type(8)));

#define WMMA_F32_F16(a, b, c) \
  __builtin_amdgcn_wmma_f32_16x16x32_f16(false, (a), false, (b), (short)0, (c), false, false)

static constexpr int Bn = 2, Sn = 2048, Dn = 2048, Hn = 16, HDn = 128, HIDn = 5888;
static constexpr float SCALEn = 0.08838834764831845f;   // 1/sqrt(128)
static constexpr float NEGn   = -65504.0f;

// ---------------- Tensor Data Mover (gfx1250) support detection ----------------
// ROCm 7.2 (clang-22): 5-arg builtin.  amdgpu-toolchain (clang-23 + therock
// headers, which ship amd_gfx1250_TDM.h): 6-arg builtin.
#if defined(__HIP_DEVICE_COMPILE__) && __has_builtin(__builtin_amdgcn_tensor_load_to_lds) && \
    __has_builtin(__builtin_amdgcn_s_wait_tensorcnt)
#define USE_TDM 1
#if __has_include(<hip/amd_detail/amd_gfx1250_TDM.h>)
#define TDM_6ARG 1
#else
#define TDM_6ARG 0
#endif
#else
#define USE_TDM 0
#endif

#if USE_TDM
typedef unsigned int tdm_u4 __attribute__((ext_vector_type(4)));
typedef int          tdm_i8 __attribute__((ext_vector_type(8)));
typedef int          tdm_i4 __attribute__((ext_vector_type(4)));

// DMA one 128-row x 32-element f16 tile (row stride = stride_elems) from global
// into LDS at lds_byte_addr, padding +4 DWORDs after every 16 DWORDs so rows
// land on a 40-half (80B) pitch — matches As/Ws[128][40].
__device__ inline void tdm_load_tile_f16(unsigned lds_byte_addr, const _Float16* g,
                                         int stride_elems) {
  const unsigned long long ga = (unsigned long long)(uintptr_t)g;
  tdm_u4 g0;
  g0.x = 1u;                                            // count = 1 descriptor
  g0.y = lds_byte_addr;                                 // D#.lds_addr
  g0.z = (unsigned)ga;                                  // global_addr[31:0]
  g0.w = (unsigned)((ga >> 32) & 0x1ffffffu) | (2u << 30);  // addr[56:32] | type=2
  const unsigned td = 0x7fffffffu;                      // huge tensor dims (no OOB)
  tdm_i8 g1;
  // d0: workgroup_mask=0 | data_size(2B)=1<<16 | pad_enable<<20 |
  //     pad_interval(16 DW)=3<<22 | pad_amount(4 DW)=3<<25
  g1[0] = (int)((1u << 16) | (1u << 20) | (3u << 22) | (3u << 25));
  g1[1] = (int)(td << 16);                              // tensor_dim0[15:0]
  g1[2] = (int)((td >> 16) | (td << 16));               // tensor_dim0[31:16] | dim1[15:0]
  g1[3] = (int)((td >> 16) | (32u << 16));              // dim1[31:16] | tile_dim0=32
  g1[4] = (int)128;                                     // tile_dim1=128, tile_dim2=0
  g1[5] = stride_elems;                                 // tensor_dim0_stride[31:0]
  g1[6] = 0;                                            // stride[47:32] | dim1_stride lo
  g1[7] = 0;
  const tdm_i4 z4 = (tdm_i4)0;
#if TDM_6ARG
  const tdm_i8 z8 = (tdm_i8)0;
  __builtin_amdgcn_tensor_load_to_lds(g0, g1, z4, z4, z8, 0);
#else
  __builtin_amdgcn_tensor_load_to_lds(g0, g1, z4, z4, 0);
#endif
}
#endif  // USE_TDM

// ---------------- WMMA fragment loaders (contiguous k = 16*(lane/16)+i, wave32) ------------

// A matrix 16x32 f16, row-major source with leading dim `ld`.
__device__ inline v16h frag_a_16x32(const _Float16* base, int ld, int lane) {
  v16h a;
  const int row = lane & 15;
  const int ko  = (lane >> 4) * 16;
  #pragma unroll
  for (int h = 0; h < 16; ++h) {
    const int k = h + ko;
    a[h] = base[row * ld + k];
  }
  return a;
}

// B matrix 32x16 f16 from a K-major (row-major KxN) source: B[k][n] = base[k*ld + n]
__device__ inline v16h frag_b_32x16_km(const _Float16* base, int ld, int lane) {
  v16h b;
  const int n  = lane & 15;
  const int ko = (lane >> 4) * 16;
  #pragma unroll
  for (int h = 0; h < 16; ++h) b[h] = base[(ko + h) * ld + n];
  return b;
}

// B matrix 32x16 f16 from an N-major (row-major NxK, i.e. B^T) source: B[k][n] = base[n*ld + k]
__device__ inline v16h frag_b_32x16_nm(const _Float16* base, int ld, int lane) {
  v16h b;
  const int n  = lane & 15;
  const int ko = (lane >> 4) * 16;
  #pragma unroll
  for (int h = 0; h < 16; ++h) b[h] = base[n * ld + ko + h];
  return b;
}

// ---------------- weight fp32 -> f16 ------------------------------------------

__global__ __launch_bounds__(256) void cvt_f16_kernel(const float* __restrict__ in,
                                                      _Float16* __restrict__ out, int n) {
  const int i = blockIdx.x * 1024 + threadIdx.x;
  #pragma unroll
  for (int j = 0; j < 4; ++j) {
    const int idx = i + j * 256;
    if (idx < n) out[idx] = (_Float16)in[idx];
  }
}

// ---------------- LayerNorm (row of 2048) -> f16 -------------------------------

__global__ __launch_bounds__(256) void layernorm_f16_kernel(const float* __restrict__ x,
                                                            const float* __restrict__ w,
                                                            const float* __restrict__ bia,
                                                            _Float16* __restrict__ out) {
  const int row = blockIdx.x, tid = threadIdx.x;
  const float* xr = x + (size_t)row * Dn;
  __shared__ float red[8];
  float v[8];
  float s = 0.f;
  #pragma unroll
  for (int i = 0; i < 8; ++i) { v[i] = xr[tid + i * 256]; s += v[i]; }
  #pragma unroll
  for (int o = 16; o >= 1; o >>= 1) s += __shfl_xor(s, o, 32);
  if ((tid & 31) == 0) red[tid >> 5] = s;
  __syncthreads();
  float tot = 0.f;
  #pragma unroll
  for (int j = 0; j < 8; ++j) tot += red[j];
  const float mean = tot * (1.0f / Dn);
  float vs = 0.f;
  #pragma unroll
  for (int i = 0; i < 8; ++i) { const float d = v[i] - mean; vs += d * d; }
  __syncthreads();
  #pragma unroll
  for (int o = 16; o >= 1; o >>= 1) vs += __shfl_xor(vs, o, 32);
  if ((tid & 31) == 0) red[tid >> 5] = vs;
  __syncthreads();
  float vt = 0.f;
  #pragma unroll
  for (int j = 0; j < 8; ++j) vt += red[j];
  const float rstd = rsqrtf(vt * (1.0f / Dn) + 1e-5f);
  #pragma unroll
  for (int i = 0; i < 8; ++i) {
    const int c = tid + i * 256;
    out[(size_t)row * Dn + c] = (_Float16)((v[i] - mean) * rstd * w[c] + bia[c]);
  }
}

// ------------- GEMM: C(MxN) = A(MxK,f16) @ W(NxK,f16)^T + bias, WMMA f16 -------
// EPI 0: outF = acc+bias        EPI 1: outF = acc+bias+res
// EPI 2: outF += acc+bias       EPI 3: outH = gelu(acc+bias) (exact erf)

template <int EPI>
__global__ __launch_bounds__(256) void gemm_f16_wmma(const _Float16* __restrict__ A,
                                                     const _Float16* __restrict__ W,
                                                     const float* __restrict__ bias,
                                                     const float* __restrict__ res,
                                                     float* __restrict__ outF,
                                                     _Float16* __restrict__ outH,
                                                     int M, int N, int K) {
  constexpr int BM = 128, BN = 128, BK = 32, LDK = BK + 8;
  __shared__ _Float16 As[BM][LDK];
  __shared__ _Float16 Ws[BN][LDK];
  const int tid = threadIdx.x, lane = tid & 31, wave = tid >> 5;
  const int wm = (wave >> 1) * 32, wn = (wave & 1) * 64;
  const int m0 = blockIdx.y * BM, n0 = blockIdx.x * BN;

  v8f acc[2][4];
  #pragma unroll
  for (int mi = 0; mi < 2; ++mi)
    #pragma unroll
    for (int ni = 0; ni < 4; ++ni) { v8f z = {}; acc[mi][ni] = z; }

#if USE_TDM
  const unsigned ldsA = (unsigned)(uintptr_t)&As[0][0];   // low 32 bits = LDS offset
  const unsigned ldsW = (unsigned)(uintptr_t)&Ws[0][0];
#endif

  const int nk = K / BK;
  for (int kt = 0; kt < nk; ++kt) {
    const int k0 = kt * BK;
    __syncthreads();
#if USE_TDM
    if (wave == 0) {
      tdm_load_tile_f16(ldsA, &A[(size_t)m0 * K + k0], K);
      tdm_load_tile_f16(ldsW, &W[(size_t)n0 * K + k0], K);
      __builtin_amdgcn_s_wait_tensorcnt(0);
    }
#else
    #pragma unroll
    for (int j = 0; j < 2; ++j) {
      const int c = tid + j * 256;
      const int r = c >> 2, kc = (c & 3) * 8;
      *(v8h*)&As[r][kc] = *(const v8h*)&A[(size_t)(m0 + r) * K + k0 + kc];
      *(v8h*)&Ws[r][kc] = *(const v8h*)&W[(size_t)(n0 + r) * K + k0 + kc];
    }
#endif
    if (tid == 0 && kt + 1 < nk) {         // -> global_prefetch_b8
      __builtin_prefetch(&A[(size_t)m0 * K + k0 + BK], 0, 1);
      __builtin_prefetch(&W[(size_t)n0 * K + k0 + BK], 0, 1);
    }
    __syncthreads();
    v16h af[2], bf[4];
    #pragma unroll
    for (int mi = 0; mi < 2; ++mi) af[mi] = frag_a_16x32(&As[wm + mi * 16][0], LDK, lane);
    #pragma unroll
    for (int ni = 0; ni < 4; ++ni) bf[ni] = frag_b_32x16_nm(&Ws[wn + ni * 16][0], LDK, lane);
    #pragma unroll
    for (int mi = 0; mi < 2; ++mi)
      #pragma unroll
      for (int ni = 0; ni < 4; ++ni)
        acc[mi][ni] = WMMA_F32_F16(af[mi], bf[ni], acc[mi][ni]);
  }

  const int half = lane >> 4, nn = lane & 15;
  #pragma unroll
  for (int mi = 0; mi < 2; ++mi)
    #pragma unroll
    for (int ni = 0; ni < 4; ++ni) {
      const int col = n0 + wn + ni * 16 + nn;
      #pragma unroll
      for (int v = 0; v < 8; ++v) {
        const int row = m0 + wm + mi * 16 + v + 8 * half;
        const size_t idx = (size_t)row * N + col;
        const float a = acc[mi][ni][v] + bias[col];
        if (EPI == 0) outF[idx] = a;
        else if (EPI == 1) outF[idx] = a + res[idx];
        else if (EPI == 2) outF[idx] += a;
        else outH[idx] = (_Float16)(0.5f * a * (1.0f + erff(a * 0.70710678118654752f)));
      }
    }
}

// --------- QKV postprocess: rmsnorm (+rope for q/k), [B,S,H,HD]->[B,H,S,HD] f16 --

template <int MODE>  // 0: rmsnorm + rope (q,k)   2: passthrough (v)
__global__ __launch_bounds__(128) void qkv_post_kernel(const float* __restrict__ in,
                                                       const float* __restrict__ nw,
                                                       const float* __restrict__ fc,
                                                       _Float16* __restrict__ out) {
  const int d = threadIdx.x;
  const int bsh = blockIdx.x;
  const int h = bsh % Hn;
  const int s = (bsh / Hn) % Sn;
  const int bb = bsh / (Hn * Sn);
  float val = in[(size_t)bsh * HDn + d];
  __shared__ float sh[HDn];
  __shared__ float red[4];
  if (MODE == 0) {
    float ss = val * val;
    #pragma unroll
    for (int o = 16; o >= 1; o >>= 1) ss += __shfl_xor(ss, o, 32);
    if ((d & 31) == 0) red[d >> 5] = ss;
    __syncthreads();
    const float tot = red[0] + red[1] + red[2] + red[3];
    const float r = rsqrtf(tot * (1.0f / HDn) + 1e-5f);
    val = val * r * nw[d];
    sh[d] = val;
    __syncthreads();
    if (d < 32) {
      const int i = d >> 1;
      const float c  = fc[((size_t)s * 16 + i) * 2 + 0];
      const float sn = fc[((size_t)s * 16 + i) * 2 + 1];
      const float x1 = sh[i], x2 = sh[i + 16];
      val = (d & 1) ? (x2 * c + x1 * sn) : (x1 * c - x2 * sn);
    }
  }
  out[(((size_t)bb * Hn + h) * Sn + s) * HDn + d] = (_Float16)val;
}

// --------------- Flash attention: per (b,h) x 128-q-row tile, WMMA -------------

__global__ __launch_bounds__(256) void attn_flash_kernel(const _Float16* __restrict__ Q,
                                                         const _Float16* __restrict__ Kh,
                                                         const _Float16* __restrict__ V,
                                                         _Float16* __restrict__ Y) {
  constexpr int LQ = HDn + 8;  // 136 halves, 272B rows (16B aligned)
  __shared__ _Float16 Qs[128][LQ];
  __shared__ _Float16 Ks[32][LQ];
  __shared__ _Float16 Vs[32][LQ];
  __shared__ _Float16 Ps[8][16][40];  // per-wave P tile, C-layout -> A-layout relay
  const int tid = threadIdx.x, lane = tid & 31, w = tid >> 5;
  const int half = lane >> 4, nn = lane & 15;
  const int qb = blockIdx.x * 128;
  const int bh = blockIdx.y;
  const int b = bh / Hn, h = bh % Hn;
  const size_t base = (size_t)bh * Sn * HDn;

  #pragma unroll
  for (int j = 0; j < 8; ++j) {
    const int c = tid + j * 256;
    const int r = c >> 4, kc = (c & 15) * 8;
    *(v8h*)&Qs[r][kc] = *(const v8h*)&Q[base + (size_t)(qb + r) * HDn + kc];
  }
  __syncthreads();
  v16h qf[4];
  #pragma unroll
  for (int kk = 0; kk < 4; ++kk) qf[kk] = frag_a_16x32(&Qs[w * 16][kk * 32], LQ, lane);

  v8f o[8];
  float m[8], l[8];
  #pragma unroll
  for (int j = 0; j < 8; ++j) { v8f z = {}; o[j] = z; }
  #pragma unroll
  for (int v = 0; v < 8; ++v) { m[v] = -3.0e38f; l[v] = 0.f; }

  const int nkt = blockIdx.x * 4 + 4;  // causal: keys [0, qb+128)
  for (int kt = 0; kt < nkt; ++kt) {
    const int kb = kt * 32;
    __syncthreads();
    #pragma unroll
    for (int j = 0; j < 2; ++j) {
      const int c = tid + j * 256;
      const int r = c >> 4, kc = (c & 15) * 8;
      *(v8h*)&Ks[r][kc] = *(const v8h*)&Kh[base + (size_t)(kb + r) * HDn + kc];
      *(v8h*)&Vs[r][kc] = *(const v8h*)&V[base + (size_t)(kb + r) * HDn + kc];
    }
    __syncthreads();

    v8f sc0 = {}, sc1 = {};
    #pragma unroll
    for (int kk = 0; kk < 4; ++kk) {
      const v16h b0 = frag_b_32x16_nm(&Ks[0][kk * 32], LQ, lane);
      const v16h b1 = frag_b_32x16_nm(&Ks[16][kk * 32], LQ, lane);
      sc0 = WMMA_F32_F16(qf[kk], b0, sc0);
      sc1 = WMMA_F32_F16(qf[kk], b1, sc1);
    }
    // scale + causal mask
    #pragma unroll
    for (int v = 0; v < 8; ++v) {
      const int q_idx = qb + w * 16 + v + 8 * half;
      const float s0 = sc0[v] * SCALEn, s1 = sc1[v] * SCALEn;
      sc0[v] = (kb + nn > q_idx) ? NEGn : s0;
      sc1[v] = (kb + 16 + nn > q_idx) ? NEGn : s1;
    }
    // online softmax: row max/sum via 16-lane xor shuffles (rows live per half)
    float alpha[8];
    #pragma unroll
    for (int v = 0; v < 8; ++v) {
      float rm = fmaxf(sc0[v], sc1[v]);
      #pragma unroll
      for (int o2 = 8; o2 >= 1; o2 >>= 1) rm = fmaxf(rm, __shfl_xor(rm, o2, 32));
      const float mn = fmaxf(m[v], rm);
      alpha[v] = __expf(m[v] - mn);
      m[v] = mn;
    }
    #pragma unroll
    for (int v = 0; v < 8; ++v) {
      const float p0 = __expf(sc0[v] - m[v]);
      const float p1 = __expf(sc1[v] - m[v]);
      sc0[v] = p0; sc1[v] = p1;
      float rs = p0 + p1;
      #pragma unroll
      for (int o2 = 8; o2 >= 1; o2 >>= 1) rs += __shfl_xor(rs, o2, 32);
      l[v] = l[v] * alpha[v] + rs;
    }
    #pragma unroll
    for (int j = 0; j < 8; ++j)
      #pragma unroll
      for (int v = 0; v < 8; ++v) o[j][v] *= alpha[v];
    // relay P through per-wave LDS: C layout -> A layout (wave-local, DS in-order)
    #pragma unroll
    for (int v = 0; v < 8; ++v) {
      Ps[w][v + 8 * half][nn]      = (_Float16)sc0[v];
      Ps[w][v + 8 * half][16 + nn] = (_Float16)sc1[v];
    }
    const v16h pf = frag_a_16x32(&Ps[w][0][0], 40, lane);
    #pragma unroll
    for (int j = 0; j < 8; ++j) {
      const v16h vf = frag_b_32x16_km(&Vs[0][j * 16], LQ, lane);
      o[j] = WMMA_F32_F16(pf, vf, o[j]);
    }
  }

  #pragma unroll
  for (int v = 0; v < 8; ++v) l[v] = 1.0f / l[v];
  #pragma unroll
  for (int j = 0; j < 8; ++j)
    #pragma unroll
    for (int v = 0; v < 8; ++v) {
      const int qs = qb + w * 16 + v + 8 * half;
      const int col = j * 16 + nn;
      Y[((size_t)(b * Sn + qs)) * Dn + h * HDn + col] = (_Float16)(o[j][v] * l[v]);
    }
}

// -------------------------------- launcher ------------------------------------

extern "C" void kernel_launch(void* const* d_in, const int* in_sizes, int n_in,
                              void* d_out, int out_size, void* d_ws, size_t ws_size,
                              hipStream_t stream) {
  (void)in_sizes; (void)n_in; (void)out_size; (void)ws_size;
  const float* x0 = (const float*)d_in[0];
  const float* x1 = (const float*)d_in[1];
  const float* x2 = (const float*)d_in[2];
  const float* x3 = (const float*)d_in[3];
  const float* freqs = (const float*)d_in[5];
  const float* wq_w = (const float*)d_in[7];  const float* wq_b = (const float*)d_in[8];
  const float* wk_w = (const float*)d_in[9];  const float* wk_b = (const float*)d_in[10];
  const float* wv_w = (const float*)d_in[11]; const float* wv_b = (const float*)d_in[12];
  const float* wo_w = (const float*)d_in[13]; const float* wo_b = (const float*)d_in[14];
  const float* qn_w = (const float*)d_in[15]; const float* kn_w = (const float*)d_in[16];
  const float* ln0_w = (const float*)d_in[17]; const float* ln0_b = (const float*)d_in[18];
  const float* ln1_w = (const float*)d_in[19]; const float* ln1_b = (const float*)d_in[20];
  const float* ln2_w = (const float*)d_in[21]; const float* ln2_b = (const float*)d_in[22];
  const float* fln_w = (const float*)d_in[23]; const float* fln_b = (const float*)d_in[24];
  const float* w1_w = (const float*)d_in[25]; const float* w1_b = (const float*)d_in[26];
  const float* w2_w = (const float*)d_in[27]; const float* w2_b = (const float*)d_in[28];
  float* out = (float*)d_out;

  const int M = Bn * Sn;  // 4096
  char* p = (char*)d_ws;
  auto take = [&](size_t bytes) { char* r = p; p += (bytes + 255) & ~(size_t)255; return r; };
  _Float16* wq16 = (_Float16*)take((size_t)Dn * Dn * 2);
  _Float16* wk16 = (_Float16*)take((size_t)Dn * Dn * 2);
  _Float16* wv16 = (_Float16*)take((size_t)Dn * Dn * 2);
  _Float16* wo16 = (_Float16*)take((size_t)Dn * Dn * 2);
  _Float16* w116 = (_Float16*)take((size_t)HIDn * Dn * 2);
  _Float16* w216 = (_Float16*)take((size_t)HIDn * Dn * 2);
  _Float16* nbuf = (_Float16*)take((size_t)M * Dn * 2);     // LN outputs (reused)
  float*    proj = (float*)take((size_t)M * Dn * 4);        // QKV pre-norm (reused)
  _Float16* qh   = (_Float16*)take((size_t)M * Dn * 2);     // [B,H,S,HD]
  _Float16* kh   = (_Float16*)take((size_t)M * Dn * 2);
  _Float16* vh   = (_Float16*)take((size_t)M * Dn * 2);
  _Float16* y16  = (_Float16*)take((size_t)M * Dn * 2);     // attention out [B,S,D]
  _Float16* g1   = (_Float16*)take((size_t)M * HIDn * 2);   // gelu(ffn1) f16

  auto cvt = [&](const float* src, _Float16* dst, int n) {
    cvt_f16_kernel<<<dim3((n + 1023) / 1024), dim3(256), 0, stream>>>(src, dst, n);
  };
  cvt(wq_w, wq16, Dn * Dn);
  cvt(wk_w, wk16, Dn * Dn);
  cvt(wv_w, wv16, Dn * Dn);
  cvt(wo_w, wo16, Dn * Dn);
  cvt(w1_w, w116, HIDn * Dn);
  cvt(w2_w, w216, HIDn * Dn);

  const dim3 blk(256);
  const dim3 gD(Dn / 128, M / 128);     // 16 x 32
  const dim3 gH(HIDn / 128, M / 128);   // 46 x 32

  // ---- Q ----
  layernorm_f16_kernel<<<M, blk, 0, stream>>>(x0, ln0_w, ln0_b, nbuf);
  gemm_f16_wmma<0><<<gD, blk, 0, stream>>>(nbuf, wq16, wq_b, nullptr, proj, nullptr, M, Dn, Dn);
  qkv_post_kernel<0><<<Bn * Sn * Hn, 128, 0, stream>>>(proj, qn_w, freqs, qh);
  // ---- K ----
  layernorm_f16_kernel<<<M, blk, 0, stream>>>(x1, ln1_w, ln1_b, nbuf);
  gemm_f16_wmma<0><<<gD, blk, 0, stream>>>(nbuf, wk16, wk_b, nullptr, proj, nullptr, M, Dn, Dn);
  qkv_post_kernel<0><<<Bn * Sn * Hn, 128, 0, stream>>>(proj, kn_w, freqs, kh);
  // ---- V ----
  layernorm_f16_kernel<<<M, blk, 0, stream>>>(x2, ln2_w, ln2_b, nbuf);
  gemm_f16_wmma<0><<<gD, blk, 0, stream>>>(nbuf, wv16, wv_b, nullptr, proj, nullptr, M, Dn, Dn);
  qkv_post_kernel<2><<<Bn * Sn * Hn, 128, 0, stream>>>(proj, nullptr, nullptr, vh);
  // ---- attention ----
  attn_flash_kernel<<<dim3(Sn / 128, Bn * Hn), blk, 0, stream>>>(qh, kh, vh, y16);
  // ---- out-proj + residual: out = x3 + y @ wo^T + wo_b ----
  gemm_f16_wmma<1><<<gD, blk, 0, stream>>>(y16, wo16, wo_b, x3, out, nullptr, M, Dn, Dn);
  // ---- FFN: out += gelu(LN(x3) @ w1^T + b1) @ w2^T + b2 ----
  layernorm_f16_kernel<<<M, blk, 0, stream>>>(x3, fln_w, fln_b, nbuf);
  gemm_f16_wmma<3><<<gH, blk, 0, stream>>>(nbuf, w116, w1_b, nullptr, nullptr, g1, M, HIDn, Dn);
  gemm_f16_wmma<2><<<gD, blk, 0, stream>>>(g1, w216, w2_b, nullptr, out, nullptr, M, Dn, HIDn);
}
